// PropagationLayer_71811853189801
// MI455X (gfx1250) — matrix operand order repaired and checked by reading, not compiled, not run on hardware
//
#include <hip/hip_runtime.h>
#include <hip/hip_bf16.h>
#include <math.h>

#define NN 8192
#define DD 256
#define HH 128

typedef _Float16 half8  __attribute__((ext_vector_type(8)));
typedef _Float16 half16 __attribute__((ext_vector_type(16)));
typedef float    float8 __attribute__((ext_vector_type(8)));

union H16 { half16 v; half8 h[2]; };

// ---------------------------------------------------------------------------
// Kernel 1: Q/K projections + cosine normalization -> f16  [N, H]
// One block per node row; thread t computes head-dim t for both Q and K.
// ---------------------------------------------------------------------------
__global__ __launch_bounds__(128) void proj_norm_kernel(
    const float* __restrict__ features, const float* __restrict__ Wq,
    const float* __restrict__ Wk, _Float16* __restrict__ Qh,
    _Float16* __restrict__ Kh)
{
    __shared__ float frow[DD];
    __shared__ float red[128];
    const int i = blockIdx.x;
    const int t = threadIdx.x;          // 0..127  == H index

    for (int d = t; d < DD; d += 128) frow[d] = features[i * DD + d];
    __syncthreads();

    float q = 0.f, k = 0.f;
    const float* wq = Wq + t * DD;
    const float* wk = Wk + t * DD;
    #pragma unroll 4
    for (int d = 0; d < DD; ++d) {
        const float f = frow[d];
        q = fmaf(f, wq[d], q);
        k = fmaf(f, wk[d], k);
    }

    // ||q|| over 128 threads
    red[t] = q * q; __syncthreads();
    for (int s = 64; s > 0; s >>= 1) { if (t < s) red[t] += red[t + s]; __syncthreads(); }
    const float qn = sqrtf(red[0]);
    __syncthreads();
    // ||k||
    red[t] = k * k; __syncthreads();
    for (int s = 64; s > 0; s >>= 1) { if (t < s) red[t] += red[t + s]; __syncthreads(); }
    const float kn = sqrtf(red[0]);

    Qh[i * HH + t] = (_Float16)(q / fmaxf(qn, 1e-4f));
    Kh[i * HH + t] = (_Float16)(k / fmaxf(kn, 1e-4f));
}

// ---------------------------------------------------------------------------
// Kernel 2: features f32 [N][D]  ->  Ft f16 [D][N]  (transposed, for B-frags)
// ---------------------------------------------------------------------------
__global__ __launch_bounds__(256) void transpose_f16_kernel(
    const float* __restrict__ features, _Float16* __restrict__ Ft)
{
    __shared__ float tile[32][33];
    const int bx = blockIdx.x;              // D/32 tiles
    const int by = blockIdx.y;              // N/32 tiles
    const int tx = threadIdx.x & 31;
    const int ty = threadIdx.x >> 5;        // 0..7
    #pragma unroll
    for (int r = 0; r < 4; ++r) {
        const int row = by * 32 + ty + r * 8;
        tile[ty + r * 8][tx] = features[row * DD + bx * 32 + tx];
    }
    __syncthreads();
    #pragma unroll
    for (int r = 0; r < 4; ++r) {
        const int dcol = bx * 32 + ty + r * 8;
        Ft[(size_t)dcol * NN + by * 32 + tx] = (_Float16)tile[tx][ty + r * 8];
    }
}

// ---------------------------------------------------------------------------
// Kernel 3: fused masked-softmax attention + propagation + keep-ratio mix.
// 8 waves / block, each wave owns a 16-row output tile. wave32 WMMA.
// ---------------------------------------------------------------------------
__global__ __launch_bounds__(256) void prop_attn_kernel(
    const float* __restrict__ features, const int* __restrict__ adj,
    const _Float16* __restrict__ Kh, const _Float16* __restrict__ Qh,
    const _Float16* __restrict__ Ft, float* __restrict__ out)
{
    __shared__ _Float16 Pbuf[8][16][32];   // per-wave P staging (C->A relayout)

    const int lane = threadIdx.x & 31;
    const int wave = threadIdx.x >> 5;
    const int n    = lane & 15;            // column-in-tile / M-in-A-frag
    const int h    = lane >> 4;            // lane half
    const int i0   = (blockIdx.x * 8 + wave) * 16;

    // A-fragments of Kn (16 rows x 128), 4 chunks of K=32, kept resident.
    // 16-bit A layout: lane(m=n,half=h) holds K = {32c+8h+0..7, 32c+16+8h+0..7}
    H16 aK[4];
    #pragma unroll
    for (int c = 0; c < 4; ++c) {
        const _Float16* base = Kh + (size_t)(i0 + n) * HH + 32 * c + 8 * h;
        aK[c].h[0] = *(const half8*)(base);
        aK[c].h[1] = *(const half8*)(base + 16);
    }

    float8 acc[16];
    #pragma unroll
    for (int c = 0; c < 16; ++c) acc[c] = (float8){};
    float rs[8];
    #pragma unroll
    for (int r = 0; r < 8; ++r) rs[r] = 0.f;

    for (int j0 = 0; j0 < NN; j0 += 32) {
        // stream next adj tile toward the caches (global_prefetch_b8)
        if (j0 + 512 < NN)
            __builtin_prefetch(&adj[(size_t)(i0 + n) * NN + j0 + 512 + h * 16], 0, 0);

        #pragma unroll
        for (int t = 0; t < 2; ++t) {
            const int jt = j0 + 16 * t;
            // B-frag of Qn^T: lane(n,h) holds Qn[jt+n][32c + {8h+0..7, 16+8h+0..7}]
            float8 s = (float8){};
            #pragma unroll
            for (int c = 0; c < 4; ++c) {
                H16 bQ;
                const _Float16* qb = Qh + (size_t)(jt + n) * HH + 32 * c + 8 * h;
                bQ.h[0] = *(const half8*)(qb);
                bQ.h[1] = *(const half8*)(qb + 16);
                s = __builtin_amdgcn_wmma_f32_16x16x32_f16(
                        false, aK[c].v, false, bQ.v, (short)0, s, false, false);
            }
            // mask + exp (scores are cosines in [-1,1] -> no running max needed)
            #pragma unroll
            for (int r = 0; r < 8; ++r) {
                const int row = i0 + r + 8 * h;           // C layout row
                const int a = adj[(size_t)row * NN + jt + n];
                const float p = (a > 0) ? __expf(s[r] - 1.0f) : 0.0f;
                rs[r] += p;
                Pbuf[wave][r + 8 * h][16 * t + n] = (_Float16)p;
            }
        }

        // Re-load P as an A-fragment (wave-private LDS, DS ops are in-order)
        H16 aP;
        aP.h[0] = *(const half8*)&Pbuf[wave][n][8 * h];
        aP.h[1] = *(const half8*)&Pbuf[wave][n][16 + 8 * h];

        // acc(16x256) += P(16x32) x F(32x256); B-frag from transposed Ft:
        // lane(n,h) holds Ft[c*16+n][j0 + {8h+0..7, 16+8h+0..7}]
        #pragma unroll
        for (int c = 0; c < 16; ++c) {
            H16 bF;
            const _Float16* fb = Ft + (size_t)(c * 16 + n) * NN + j0 + 8 * h;
            bF.h[0] = *(const half8*)(fb);
            bF.h[1] = *(const half8*)(fb + 16);
            acc[c] = __builtin_amdgcn_wmma_f32_16x16x32_f16(
                        false, aP.v, false, bF.v, (short)0, acc[c], false, false);
        }
    }

    // Row sums: reduce across the 16 lanes of each half (wave32 -> width 16)
    #pragma unroll
    for (int r = 0; r < 8; ++r) {
        float v = rs[r];
        v += __shfl_xor(v, 8, 16);
        v += __shfl_xor(v, 4, 16);
        v += __shfl_xor(v, 2, 16);
        v += __shfl_xor(v, 1, 16);
        rs[r] = v;
    }

    // Epilogue: normalize, no-neighbor fixup, keep-ratio mix (all f32)
    #pragma unroll
    for (int r = 0; r < 8; ++r) {
        const int row = i0 + r + 8 * h;
        const float l = rs[r];
        const float inv = (l > 0.0f) ? (1.0f / l) : 0.0f;
        #pragma unroll
        for (int c = 0; c < 16; ++c) {
            const int col = c * 16 + n;
            const float f = features[(size_t)row * DD + col];
            const float v = (l > 0.0f) ? (0.5f * acc[c][r] * inv + 0.5f * f) : f;
            out[(size_t)row * DD + col] = v;
        }
    }
}

// ---------------------------------------------------------------------------
extern "C" void kernel_launch(void* const* d_in, const int* in_sizes, int n_in,
                              void* d_out, int out_size, void* d_ws, size_t ws_size,
                              hipStream_t stream) {
    const float* features = (const float*)d_in[0];
    const int*   adj      = (const int*)d_in[1];
    const float* Wq       = (const float*)d_in[2];
    const float* Wk       = (const float*)d_in[3];
    float*       out      = (float*)d_out;

    char* ws = (char*)d_ws;
    _Float16* Kh = (_Float16*)(ws);                                 // 2 MB
    _Float16* Qh = (_Float16*)(ws + (size_t)NN * HH * 2);           // 2 MB
    _Float16* Ft = (_Float16*)(ws + (size_t)NN * HH * 4);           // 4 MB

    proj_norm_kernel<<<NN, 128, 0, stream>>>(features, Wq, Wk, Qh, Kh);
    transpose_f16_kernel<<<dim3(DD / 32, NN / 32), 256, 0, stream>>>(features, Ft);
    prop_attn_kernel<<<NN / 128, 256, 0, stream>>>(features, adj, Kh, Qh, Ft, out);
}
